// GraphSpatialBlock_74801150427321
// MI455X (gfx1250) — matrix operand order repaired and checked by reading, not compiled, not run on hardware
//
#include <hip/hip_runtime.h>

// ---------------------------------------------------------------------------
// GraphSpatialBlock for MI455X (gfx1250, wave32, WMMA) -- single fused kernel.
//   x[B=8,C=128,N=64,T=512] --(W: C->H*D=128)--> GAT(64 nodes, 320 edges)
//   --> node-mean --> BatchNorm(eval) --> y[B,128,1,T]
// One block = 2 bt-slices x all 64 nodes. h lives entirely in LDS: global
// traffic is x (read once; also fits in the 192MB L2) + y + tiny CSR
// (~130 MB => ~5.6us at 23.3 TB/s). GEMM on v_wmma_f32_16x16x32_bf16 with
// ds_load_b128 fragment feeds; x staged as pipelined float2 (global_load_b64)
// batches; CSR staged via global_load_async_to_lds_b32 (ASYNCcnt) overlapped
// with the GEMM phase.
// ---------------------------------------------------------------------------

typedef __attribute__((ext_vector_type(16))) __bf16 v16bf;
typedef __attribute__((ext_vector_type(8)))  __bf16 v8bf;
typedef __attribute__((ext_vector_type(8)))  float  v8f;

#define B_BATCH 8
#define C_IN    128
#define N_NODES 64
#define T_STEPS 512
#define BT      (B_BATCH * T_STEPS)   // 4096
#define H_HEADS 4
#define D_HEAD  32
#define OUT_CH  (H_HEADS * D_HEAD)    // 128
#define E_EDGES 256
#define EP      (E_EDGES + N_NODES)   // 320 edges incl. self loops
#define G_SLICE 2                     // bt slices per block; M = G*N = 128
#define LDA     40                    // A row stride (bf16): 80 B, 16B aligned
#define LDB     136                   // W^T row stride (bf16): 272 B, 16B aligned
#define HLD     132                   // h LDS row stride (bf16)

__device__ __forceinline__ __bf16 f2bf(float f) {
  union { float f; unsigned u; } v; v.f = f;
  unsigned r = (v.u + 0x7fffu + ((v.u >> 16) & 1u)) >> 16;   // RNE
  union { unsigned short s; __bf16 b; } o; o.s = (unsigned short)r;
  return o.b;
}

// async copy of one dword global -> LDS (per active lane); tracked by ASYNCcnt
__device__ __forceinline__ void async_copy_b32(void* lds_ptr, const void* gptr) {
  unsigned lds_off = (unsigned)(unsigned long long)lds_ptr;   // LDS aperture low bits
  unsigned long long ga = (unsigned long long)gptr;
  asm volatile("global_load_async_to_lds_b32 %0, %1, off"
               :: "v"(lds_off), "v"(ga) : "memory");
}

// ---------------------------------------------------------------------------
// Fused kernel. grid = BT/G_SLICE = 2048, block = 256 (8 waves).
// M rows are (node n, slice tl) pairs: row = n*G_SLICE + tl.
// ---------------------------------------------------------------------------
__global__ __launch_bounds__(256)
void k_fused_gat(const float* __restrict__ x, const float* __restrict__ W,
                 const float* __restrict__ att_src, const float* __restrict__ att_dst,
                 const int* __restrict__ ptr_ws, const int* __restrict__ col_ws,
                 const float* __restrict__ bias,
                 const float* __restrict__ gamma, const float* __restrict__ beta,
                 const float* __restrict__ mean,  const float* __restrict__ var,
                 float* __restrict__ out) {
  __shared__ __bf16 sBT[OUT_CH * LDB];            // W^T: [col][k]      34816 B
  __shared__ __bf16 sA[128 * LDA];                // A k-chunk          10240 B
  __shared__ __bf16 hL[128 * HLD];                // h tile (all nodes) 33792 B
  __shared__ float  saL[128 * H_HEADS];           // a_src per (row,h)   2048 B
  __shared__ float  sdL[128 * H_HEADS];           // a_dst per (row,h)   2048 B
  __shared__ int    sp[N_NODES + 1];              // CSR row ptr
  __shared__ int    sc[EP];                       // CSR src ids

  const int tid = threadIdx.x;
  const int bt0 = blockIdx.x * G_SLICE;
  const int b   = bt0 >> 9;                       // bt0 / T
  const int t0  = bt0 & (T_STEPS - 1);            // even; G divides T -> same b

  // Kick off async CSR staging now; consumed after the GEMM phase.
  for (int i = tid; i <= N_NODES; i += 256) async_copy_b32(&sp[i], ptr_ws + i);
  for (int i = tid; i < EP; i += 256)       async_copy_b32(&sc[i], col_ws + i);

  // Stage W^T -> LDS once (coalesced global read, f32 -> bf16)
  for (int i = tid; i < C_IN * OUT_CH; i += 256) {
    const int col = i & 127, k = i >> 7;
    sBT[col * LDB + k] = f2bf(W[i]);
  }

  const int wave = tid >> 5, lane = tid & 31;
  const int g = lane >> 4, l16 = lane & 15;
  const int m0 = wave * 16;                       // one 16-row strip per wave

  // Per-lane attention vector values for the fused dot epilogue
  float av[8], dv[8];
#pragma unroll
  for (int nt = 0; nt < 8; ++nt) {
    av[nt] = att_src[nt * 16 + l16];
    dv[nt] = att_dst[nt * 16 + l16];
  }

  v8f acc[8] = {};                                // 8 N-tiles of 16x16 f32

  // ---- GEMM phase: h[row,:] = sum_c x[b,c,n,t0+tl] * W[c,:] ----
  const float* xbase = x + ((size_t)b * C_IN) * N_NODES * T_STEPS + t0;
  const size_t cstride = (size_t)N_NODES * T_STEPS;

  for (int kc = 0; kc < 4; ++kc) {                // K = 128 in chunks of 32
    // Pipelined x staging: 8 float2 loads in flight per thread, then convert.
    // idx = (chunk-local channel cl)*64 + node n; float2 spans tl = 0,1.
    float2 xv[8];
#pragma unroll
    for (int j = 0; j < 8; ++j) {
      const int idx = tid + 256 * j;
      const int n = idx & 63, cl = idx >> 6;
      const float* p = xbase + (size_t)(kc * 32 + cl) * cstride + n * T_STEPS;
      xv[j] = *(const float2*)p;
      if (kc < 3) __builtin_prefetch(p + 32 * cstride, 0, 1);   // next K chunk
    }
    __syncthreads();   // previous iteration's fragment reads done
#pragma unroll
    for (int j = 0; j < 8; ++j) {
      const int idx = tid + 256 * j;
      const int n = idx & 63, cl = idx >> 6;
      sA[(n * G_SLICE + 0) * LDA + cl] = f2bf(xv[j].x);
      sA[(n * G_SLICE + 1) * LDA + cl] = f2bf(xv[j].y);
    }
    __syncthreads();

    // A fragment: lane row M = l16; K runs 8g..8g+7 and 16+8g..16+8g+7
    // -> two contiguous 16B LDS loads (ds_load_b128)
    const __bf16* arow = sA + (m0 + l16) * LDA;
    const v8bf a_lo = *(const v8bf*)(arow + 8 * g);
    const v8bf a_hi = *(const v8bf*)(arow + 16 + 8 * g);
    const v16bf aF = __builtin_shufflevector(a_lo, a_hi,
        0, 1, 2, 3, 4, 5, 6, 7, 8, 9, 10, 11, 12, 13, 14, 15);

#pragma unroll
    for (int nt = 0; nt < 8; ++nt) {
      // B fragment: lane col N = l16; K = kc*32 + 8g..8g+7 and 16+8g..16+8g+7 in W^T
      const __bf16* brow = sBT + (nt * 16 + l16) * LDB + kc * 32;
      const v8bf b_lo = *(const v8bf*)(brow + 8 * g);
      const v8bf b_hi = *(const v8bf*)(brow + 16 + 8 * g);
      const v16bf bF = __builtin_shufflevector(b_lo, b_hi,
          0, 1, 2, 3, 4, 5, 6, 7, 8, 9, 10, 11, 12, 13, 14, 15);
      acc[nt] = __builtin_amdgcn_wmma_f32_16x16x32_bf16(
          false, aF, false, bF, (short)0, acc[nt], false, false);
    }
  }

  // h tile -> LDS. C/D layout: VGPR r -> row M = r + 8*g, col N = nt*16 + l16
#pragma unroll
  for (int nt = 0; nt < 8; ++nt) {
#pragma unroll
    for (int r = 0; r < 8; ++r) {
      hL[(m0 + r + (g << 3)) * HLD + nt * 16 + l16] = f2bf(acc[nt][r]);
    }
  }

  // Fused attention dots: head hh = N-tiles {2hh,2hh+1}; 16-lane shfl_xor
  // butterfly per half-wave, result stored to LDS.
#pragma unroll
  for (int hh = 0; hh < H_HEADS; ++hh) {
#pragma unroll
    for (int r = 0; r < 8; ++r) {
      float ps = acc[2 * hh][r] * av[2 * hh] + acc[2 * hh + 1][r] * av[2 * hh + 1];
      float pd = acc[2 * hh][r] * dv[2 * hh] + acc[2 * hh + 1][r] * dv[2 * hh + 1];
#pragma unroll
      for (int msk = 8; msk >= 1; msk >>= 1) {
        ps += __shfl_xor(ps, msk, 32);
        pd += __shfl_xor(pd, msk, 32);
      }
      if (l16 == 0) {
        const int row = m0 + (g << 3) + r;
        saL[row * H_HEADS + hh] = ps;
        sdL[row * H_HEADS + hh] = pd;
      }
    }
  }

  asm volatile("s_wait_asynccnt 0x0" ::: "memory");   // CSR tables landed
  __syncthreads();

  // ---- Aggregation phase: wave = (tl, head), lane = d ----
  const int tl = wave & (G_SLICE - 1);
  const int hh = wave >> 1;
  const int d  = lane;

  float accv = 0.f;   // mean-over-nodes accumulator for channel (hh,d), slice tl
  for (int n = 0; n < N_NODES; ++n) {
    const int p0 = sp[n], p1 = sp[n + 1];
    const float ad = sdL[(n * G_SLICE + tl) * H_HEADS + hh];
    float mx = -1e30f;
    for (int p = p0; p < p1; ++p) {
      float s = saL[(sc[p] * G_SLICE + tl) * H_HEADS + hh] + ad;
      s = (s > 0.f) ? s : 0.2f * s;               // leaky_relu(0.2)
      mx = fmaxf(mx, s);
    }
    float den = 0.f;
    for (int p = p0; p < p1; ++p) {
      float s = saL[(sc[p] * G_SLICE + tl) * H_HEADS + hh] + ad;
      s = (s > 0.f) ? s : 0.2f * s;
      den += __expf(s - mx);
    }
    const float inv = 1.f / den;
    for (int p = p0; p < p1; ++p) {
      const int src = sc[p];
      float s = saL[(src * G_SLICE + tl) * H_HEADS + hh] + ad;
      s = (s > 0.f) ? s : 0.2f * s;
      const float w = __expf(s - mx) * inv;
      accv += w * (float)hL[(src * G_SLICE + tl) * HLD + hh * D_HEAD + d];
    }
  }

  const int oc = hh * D_HEAD + d;
  const float val = accv * (1.f / N_NODES) + bias[oc];
  const float y = (val - mean[oc]) * rsqrtf(var[oc] + 1e-5f) * gamma[oc] + beta[oc];
  out[((size_t)(b * OUT_CH + oc)) * T_STEPS + t0 + tl] = y;   // y[B,128,1,T]
}

// ---------------------------------------------------------------------------
// CSR builder: dst-sorted edge list (+ self loops). One tiny block.
// ---------------------------------------------------------------------------
__global__ void k_build_csr(const int* __restrict__ ei,
                            int* __restrict__ ptr_ws, int* __restrict__ col_ws) {
  __shared__ int cnt[N_NODES], spp[N_NODES + 1], cur[N_NODES];
  const int tid = threadIdx.x;   // 512 threads
  if (tid < N_NODES) cnt[tid] = 0;
  __syncthreads();
  int src = 0, dst = 0;
  if (tid < EP) {
    if (tid < E_EDGES) { src = ei[tid]; dst = ei[E_EDGES + tid]; }
    else               { src = tid - E_EDGES; dst = src; }       // self loop
    atomicAdd(&cnt[dst], 1);
  }
  __syncthreads();
  if (tid == 0) {
    int run = 0;
    for (int i = 0; i < N_NODES; ++i) { spp[i] = run; run += cnt[i]; }
    spp[N_NODES] = run;
  }
  __syncthreads();
  if (tid < N_NODES) cur[tid] = spp[tid];
  __syncthreads();
  if (tid < EP) {
    const int pos = atomicAdd(&cur[dst], 1);
    col_ws[pos] = src;
  }
  if (tid <= N_NODES) ptr_ws[tid] = spp[tid];
}

// ---------------------------------------------------------------------------
extern "C" void kernel_launch(void* const* d_in, const int* in_sizes, int n_in,
                              void* d_out, int out_size, void* d_ws, size_t ws_size,
                              hipStream_t stream) {
  const float* x       = (const float*)d_in[0];
  const float* W       = (const float*)d_in[1];
  const float* att_src = (const float*)d_in[2];
  const float* att_dst = (const float*)d_in[3];
  const float* bias    = (const float*)d_in[4];
  const float* gamma   = (const float*)d_in[5];
  const float* beta    = (const float*)d_in[6];
  const float* mean    = (const float*)d_in[7];
  const float* var     = (const float*)d_in[8];
  const int*   ei      = (const int*)d_in[9];
  float* out = (float*)d_out;

  // Workspace: just the CSR (h / a_src / a_dst live in LDS).
  int* ptr_ws = (int*)d_ws;
  int* col_ws = ptr_ws + 128;

  k_build_csr<<<1, 512, 0, stream>>>(ei, ptr_ws, col_ws);
  k_fused_gat<<<BT / G_SLICE, 256, 0, stream>>>(
      x, W, att_src, att_dst, ptr_ws, col_ws,
      bias, gamma, beta, mean, var, out);
}
